// PaddedModel_71519795413525
// MI455X (gfx1250) — matrix operand order repaired and checked by reading, not compiled, not run on hardware
//
#include <hip/hip_runtime.h>

// Problem constants (match the JAX reference)
constexpr int TB  = 512;   // timesteps
constexpr int NIN = 10;    // input features
constexpr int NH  = 64;    // hidden

typedef __attribute__((ext_vector_type(16))) unsigned short v16us;
typedef __attribute__((ext_vector_type(16))) __bf16        v16bf;
typedef __attribute__((ext_vector_type(8)))  float         v8f;

struct U8x32 { uint4 a, b; };     // 32 bytes == one v16us / v16bf fragment
struct V8u   { unsigned u[8]; };  // 32-byte uint view of a fragment

__device__ __forceinline__ unsigned f2bf(float f) {
    unsigned u = __builtin_bit_cast(unsigned, f);
    u += 0x7FFFu + ((u >> 16) & 1u);          // round-to-nearest-even
    return u >> 16;
}

// Pack two floats into a bf16 pair (lo | hi<<16); RNE either way.
__device__ __forceinline__ unsigned pk_bf16(float lo, float hiv) {
#if __has_builtin(__builtin_amdgcn_cvt_pk_bf16_f32)
    typedef __attribute__((ext_vector_type(2))) __bf16 v2bf;
    v2bf v = __builtin_amdgcn_cvt_pk_bf16_f32(lo, hiv);
    return __builtin_bit_cast(unsigned, v);
#else
    // fptrunc f32->bf16: lets ISel pick native v_cvt_* if gfx1250 has them
    unsigned short a = __builtin_bit_cast(unsigned short, (__bf16)lo);
    unsigned short b = __builtin_bit_cast(unsigned short, (__bf16)hiv);
    return (unsigned)a | ((unsigned)b << 16);
#endif
}

__device__ __forceinline__ v16us ldfrag_g(const uint4* __restrict__ p) {
    U8x32 u; u.a = p[0]; u.b = p[1];
    return __builtin_bit_cast(v16us, u);
}

// A-fragment chunk pair from LDS: 16B at p, 16B at p+16 halfs
__device__ __forceinline__ v16us ldfrag_lds(const unsigned short* p) {
    U8x32 u;
    u.a = *(const uint4*)(p);
    u.b = *(const uint4*)(p + 16);
    return __builtin_bit_cast(v16us, u);
}

__device__ __forceinline__ float tanh_fast(float x) {
    // tanh(x) = 1 - 2/(exp(2x)+1); exp(2x) = exp2(x * 2*log2(e))
    float e = __builtin_amdgcn_exp2f(x * 2.885390082f);
    return 1.0f - 2.0f * __builtin_amdgcn_rcpf(e + 1.0f);
}

#define WMMA_BF16(A, Bv, C)                                                    \
    __builtin_amdgcn_wmma_f32_16x16x32_bf16(                                   \
        false, __builtin_bit_cast(v16bf, (A)),                                 \
        false, __builtin_bit_cast(v16bf, (Bv)),                                \
        (short)0, (C), false, false)

// ---------------------------------------------------------------------------
// Prep: W_hh [T,64,64] f32 -> bf16 in WMMA-B-fragment order.
// Fragment (t, kk, j): lane l holds col = j*16 + (l&15),
// K = kk*32 + (l>>4)*8 + i + (i>=8 ? 8 : 0), i=0..15 (32B per lane).
// ---------------------------------------------------------------------------
__global__ __launch_bounds__(256) void prep_whh_bf16(
    const float* __restrict__ whh, unsigned short* __restrict__ dst) {
    int tid = blockIdx.x * 256 + threadIdx.x;        // [0, T*8*32)
    int t   = tid >> 8;
    int sub = tid & 255;
    int l   = sub & 31;
    int j   = (sub >> 5) & 3;
    int kk  = sub >> 7;
    int col = j * 16 + (l & 15);
    int kb  = kk * 32 + (l >> 4) * 8;
    const float* src = whh + (size_t)t * NH * NH + col;
    v16us v;
#pragma unroll
    for (int i = 0; i < 16; ++i) v[i] = (unsigned short)f2bf(src[(size_t)(kb + i + (i >= 8 ? 8 : 0)) * NH]);
    U8x32 u = __builtin_bit_cast(U8x32, v);
    uint4* d = (uint4*)(dst + (size_t)tid * 16);
    d[0] = u.a; d[1] = u.b;
}

// Prep: W_xh [T,10,64] f32 -> bf16 B-fragments, K padded 10 -> 32 with zeros.
__global__ __launch_bounds__(256) void prep_wxh_bf16(
    const float* __restrict__ wxh, unsigned short* __restrict__ dst) {
    int tid = blockIdx.x * 256 + threadIdx.x;        // [0, T*4*32)
    int t   = tid >> 7;
    int sub = tid & 127;
    int l   = sub & 31;
    int j   = sub >> 5;
    int col = j * 16 + (l & 15);
    int kb  = (l >> 4) * 8;
    v16us v;
#pragma unroll
    for (int i = 0; i < 16; ++i) {
        int k = kb + i + (i >= 8 ? 8 : 0);
        v[i] = (k < NIN)
                 ? (unsigned short)f2bf(wxh[(size_t)t * NIN * NH + (size_t)k * NH + col])
                 : (unsigned short)0;
    }
    U8x32 u = __builtin_bit_cast(U8x32, v);
    uint4* d = (uint4*)(dst + (size_t)tid * 16);
    d[0] = u.a; d[1] = u.b;
}

// ---------------------------------------------------------------------------
// Pipeline buffer: all B-fragments + raw x for one timestep (96 + 8 VGPRs)
// ---------------------------------------------------------------------------
struct StepBuf {
    v16us bx[4];   // W_xh fragments (K 0..31, real K=10)
    v16us b0[4];   // W_hh fragments K  0..31
    v16us b1[4];   // W_hh fragments K 32..63
    float2 xr[4];  // raw x floats for this lane's A chunk
};

__device__ __forceinline__ void load_step(StepBuf& F, int t,
                                          const uint4* __restrict__ wxhW,
                                          const uint4* __restrict__ whhW,
                                          const float* __restrict__ xrow,
                                          int lane) {
    const uint4* wxt = wxhW + (size_t)t * 4 * 64 + lane * 2;
    const uint4* wht = whhW + (size_t)t * 8 * 64 + lane * 2;
#pragma unroll
    for (int j = 0; j < 4; ++j) {
        F.bx[j] = ldfrag_g(wxt + j * 64);
        F.b0[j] = ldfrag_g(wht + j * 64);
        F.b1[j] = ldfrag_g(wht + (4 + j) * 64);
    }
    const float2* xp = (const float2*)(xrow + (size_t)t * NIN);
    F.xr[0] = xp[0]; F.xr[1] = xp[1]; F.xr[2] = xp[2]; F.xr[3] = xp[3];
}

// ---------------------------------------------------------------------------
// Main: one wave owns a 16-row M-tile for all 512 steps.
// 12 x v_wmma_f32_16x16x32_bf16 per step (x-GEMM runs in parallel with the
// depth-2 h-GEMM chain, merged with packed f32 adds), one-step software
// pipeline on the weight/x loads, t+2 prefetch into the WGP cache, h recycled
// C->A layout through a wave-private padded LDS tile (stride 72 halfs:
// conflict-free b128 A-fragment reads). Branch-free lane work keeps EXEC
// all-1s for the WMMA region.
// ---------------------------------------------------------------------------
__global__ __launch_bounds__(256, 1) void rnn512_wmma(
    const float*     __restrict__ x,     // [B, T, 10]
    const long long* __restrict__ lens,  // [B]
    const uint4*     __restrict__ whhW,  // bf16 fragments
    const uint4*     __restrict__ wxhW,  // bf16 fragments
    const float*     __restrict__ wlin,  // [64]
    const float*     __restrict__ blin,  // [1]
    float*           __restrict__ out)   // [B]
{
    __shared__ unsigned short hbuf[8][16 * 72];
    const int lane = threadIdx.x & 31;
    const int wid  = threadIdx.x >> 5;
    const int tile = blockIdx.x * 8 + wid;
    const int m0   = tile * 16;
    const int m    = lane & 15;
    const int hi   = lane >> 4;
    unsigned short* hb = hbuf[wid];

    // zero the wave-private h tile (h0 = 0)
    { unsigned int* hz = (unsigned int*)hb;
#pragma unroll
      for (int i = 0; i < 18; ++i) hz[lane + 32 * i] = 0u; }
    __builtin_amdgcn_wave_barrier();

    // lengths for the 8 rows this lane owns in the C/D layout (row = r + 8*hi)
    int lenr[8];
#pragma unroll
    for (int r = 0; r < 8; ++r) lenr[r] = (int)lens[m0 + hi * 8 + r];

    float wl[4];
#pragma unroll
    for (int j = 0; j < 4; ++j) wl[j] = wlin[m + 16 * j];

    // h_prev held as packed bf16 pairs: hp[j][p] = rows (2p | 2p+1 << 16)
    unsigned hp[4][4];
#pragma unroll
    for (int j = 0; j < 4; ++j)
#pragma unroll
        for (int p = 0; p < 4; ++p) hp[j][p] = 0u;

    // per-lane x base: A layout row = m, K chunk starts at hi*8
    const float* xrow = x + ((size_t)(m0 + m) * TB) * NIN + hi * 8;

    StepBuf F;
    load_step(F, 0, wxhW, whhW, xrow, lane);

    const float hmask = hi ? 0.0f : 1.0f;   // hi lanes: only K=8,9 real

#pragma unroll 1
    for (int t = 0; t < TB; ++t) {
        // ---- A fragment: x_t (16x32 bf16, real K = 10), branch-free ----
        V8u xu;
        xu.u[0] = pk_bf16(F.xr[0].x, F.xr[0].y);
        xu.u[1] = pk_bf16(F.xr[1].x * hmask, F.xr[1].y * hmask);
        xu.u[2] = pk_bf16(F.xr[2].x * hmask, F.xr[2].y * hmask);
        xu.u[3] = pk_bf16(F.xr[3].x * hmask, F.xr[3].y * hmask);
        xu.u[4] = 0u; xu.u[5] = 0u; xu.u[6] = 0u; xu.u[7] = 0u;
        v16us xa = __builtin_bit_cast(v16us, xu);

        // ---- A fragments: h (16x64 bf16) from LDS ----
        const int ab = m * 72 + hi * 8;
        v16us hA0 = ldfrag_lds(hb + ab);        // K  0..31
        v16us hA1 = ldfrag_lds(hb + ab + 32);   // K 32..63

        // ---- 12 WMMAs: x-GEMM parallel to depth-2 h-GEMM chain ----
        v8f cz = { 0.f, 0.f, 0.f, 0.f, 0.f, 0.f, 0.f, 0.f };
        v8f acc[4];
#pragma unroll
        for (int j = 0; j < 4; ++j) {
            v8f cx = WMMA_BF16(xa,  F.bx[j], cz);
            v8f ch = WMMA_BF16(hA0, F.b0[j], cz);
            ch     = WMMA_BF16(hA1, F.b1[j], ch);
#pragma unroll
            for (int k = 0; k < 8; ++k) ch[k] += cx[k];
            acc[j] = ch;
        }

        // ---- prefetch t+2 weights into the WGP cache (full 12 KB) ----
        {
            int tp = (t + 2 < TB) ? t + 2 : TB - 1;
            const char* ph = (const char*)(whhW + (size_t)tp * 8 * 64) + lane * 32;
            const char* px = (const char*)(wxhW + (size_t)tp * 4 * 64) + lane * 32;
#pragma unroll
            for (int f = 0; f < 8; ++f) __builtin_prefetch(ph + f * 1024, 0, 3);
#pragma unroll
            for (int f = 0; f < 4; ++f) __builtin_prefetch(px + f * 1024, 0, 3);
        }

        // ---- software pipeline: pull t+1 fragments while epilogue runs ----
        {
            int tn = (t + 1 < TB) ? t + 1 : t;
            load_step(F, tn, wxhW, whhW, xrow, lane);
        }

        // ---- tanh + length-freeze + write next-step h tile ----
#pragma unroll
        for (int j = 0; j < 4; ++j) {
#pragma unroll
            for (int p = 0; p < 4; ++p) {
                const int r0 = 2 * p, r1 = 2 * p + 1;
                float th0 = tanh_fast(acc[j][r0]);
                float th1 = tanh_fast(acc[j][r1]);
                unsigned prev = hp[j][p];
                float p0 = __builtin_bit_cast(float, prev << 16);
                float p1 = __builtin_bit_cast(float, prev & 0xFFFF0000u);
                float hn0 = (t < lenr[r0]) ? th0 : p0;
                float hn1 = (t < lenr[r1]) ? th1 : p1;
                unsigned np = pk_bf16(hn0, hn1);
                hp[j][p] = np;
                hb[(r0 + 8 * hi) * 72 + m + 16 * j] = (unsigned short)np;
                hb[(r1 + 8 * hi) * 72 + m + 16 * j] = (unsigned short)(np >> 16);
            }
        }
        __builtin_amdgcn_wave_barrier();
    }

    // ---- final projection: out[b] = h[b,:] @ W_lin + b_lin (OUT = 1) ----
    float pacc[8];
#pragma unroll
    for (int r = 0; r < 8; ++r) pacc[r] = 0.0f;
#pragma unroll
    for (int j = 0; j < 4; ++j)
#pragma unroll
        for (int p = 0; p < 4; ++p) {
            unsigned v = hp[j][p];
            pacc[2 * p]     += __builtin_bit_cast(float, v << 16)          * wl[j];
            pacc[2 * p + 1] += __builtin_bit_cast(float, v & 0xFFFF0000u)  * wl[j];
        }
#pragma unroll
    for (int off = 1; off < 16; off <<= 1)
#pragma unroll
        for (int r = 0; r < 8; ++r) pacc[r] += __shfl_xor(pacc[r], off, 32);
    if (m == 0) {
        float bb = blin[0];
#pragma unroll
        for (int r = 0; r < 8; ++r) out[m0 + hi * 8 + r] = pacc[r] + bb;
    }
}

// ---------------------------------------------------------------------------
extern "C" void kernel_launch(void* const* d_in, const int* in_sizes, int n_in,
                              void* d_out, int out_size, void* d_ws, size_t ws_size,
                              hipStream_t stream) {
    (void)in_sizes; (void)n_in; (void)out_size; (void)ws_size;
    const float*     x    = (const float*)d_in[0];      // [8192,512,10]
    const long long* lens = (const long long*)d_in[1];  // [8192] int64
    const float*     wxh  = (const float*)d_in[2];      // [512,10,64]
    const float*     whh  = (const float*)d_in[3];      // [512,64,64]
    const float*     wlin = (const float*)d_in[4];      // [64,1]
    const float*     blin = (const float*)d_in[5];      // [1]

    unsigned short* whhW = (unsigned short*)d_ws;                 // 4 MB
    unsigned short* wxhW = whhW + (size_t)TB * 8 * 32 * 16;       // +2 MB

    prep_whh_bf16<<<512, 256, 0, stream>>>(whh, whhW);
    prep_wxh_bf16<<<256, 256, 0, stream>>>(wxh, wxhW);
    rnn512_wmma<<<64, 256, 0, stream>>>(x, lens, (const uint4*)whhW,
                                        (const uint4*)wxhW, wlin, blin,
                                        (float*)d_out);
}
